// MLPNetworkWithAttention2_80169859547544
// MI455X (gfx1250) — matrix operand order repaired and checked by reading, not compiled, not run on hardware
//
#include <hip/hip_runtime.h>
#include <math.h>

// ---------------------------------------------------------------------------
// Fused MLP+attention Q-network for MI455X (gfx1250, wave32, WMMA bf16).
//
// Weights are pre-packed (prep_pack) into WMMA B-operand layout in d_ws:
//   B fragment (32x16, KxN), lane l, element e:  K = (e<8 ? e : e+8) + 8*(l>=16), N = l%16
// A fragments (16x32, MxK), lane l, element e:   M = l%16,
//   K = (e<8 ? e : e+8) + 8*(l>=16)
// C/D tiles (16x16 f32), lane l, element r:      M = r + 8*(l>=16), N = l%16
// ---------------------------------------------------------------------------

typedef __attribute__((ext_vector_type(16))) __bf16 v16bf;
typedef __attribute__((ext_vector_type(8)))  float  v8f;

__device__ __forceinline__ v8f vzero() {
  v8f z;
#pragma unroll
  for (int i = 0; i < 8; ++i) z[i] = 0.0f;
  return z;
}

__device__ __forceinline__ v8f wmma_bf16(v16bf a, v16bf b, v8f c) {
  return __builtin_amdgcn_wmma_f32_16x16x32_bf16(
      /*neg_a=*/false, a, /*neg_b=*/false, b,
      /*c_mod=*/(short)0, c, /*reuse_a=*/false, /*reuse_b=*/false);
}

// Packed-B fragment load: blk = (kc*8 + nt) [heads: ((k*4+kc)*8 + nt)]
__device__ __forceinline__ v16bf ldB(const __bf16* __restrict__ p, int blk, int lane) {
  return *(const v16bf*)(p + (size_t)blk * 512 + lane * 16);
}

// -------------------- workspace layout (bf16 elements) ---------------------
// P_OBS   @ 0       : 1  kc * 8 nt * 512 =   4096   (Wobs  [128,12], K pad->32)
// P_ENC   @ 4096    : 16 kc * 8 nt * 512 =  65536   (Wenc  [128,512])
// P_HEADS @ 69632   : 32 kb * 8 nt * 512 = 131072   (Wheads[8,128,128], kb=k*4+kc)
// P_DEC   @ 200704  : 8  kc * 8 nt * 512 =  32768   (Wdec  [128,256])
// total 233472 bf16 = 466944 bytes
#define PK_OBS   0
#define PK_ENC   4096
#define PK_HEADS 69632
#define PK_DEC   200704
#define PK_TOTAL 233472

__global__ void prep_pack(const float* __restrict__ Wobs,
                          const float* __restrict__ Wenc,
                          const float* __restrict__ Wheads,
                          const float* __restrict__ Wdec,
                          __bf16* __restrict__ ws) {
  int idx = blockIdx.x * blockDim.x + threadIdx.x;
  if (idx >= PK_TOTAL) return;
  float val = 0.0f;
  if (idx < PK_ENC) {                      // obs, kc == 0
    int local = idx - PK_OBS;
    int blk = local >> 9, within = local & 511;
    int lane = within >> 4, e = within & 15;
    int nt = blk & 7;
    int N = nt * 16 + (lane & 15);
    int K = (e < 8 ? e : e + 8) + ((lane >> 4) << 3);
    val = (K < 12) ? Wobs[N * 12 + K] : 0.0f;
  } else if (idx < PK_HEADS) {             // enc
    int local = idx - PK_ENC;
    int blk = local >> 9, within = local & 511;
    int lane = within >> 4, e = within & 15;
    int nt = blk & 7, kc = blk >> 3;
    int N = nt * 16 + (lane & 15);
    int K = kc * 32 + (e < 8 ? e : e + 8) + ((lane >> 4) << 3);
    val = Wenc[N * 512 + K];
  } else if (idx < PK_DEC) {               // heads
    int local = idx - PK_HEADS;
    int blk = local >> 9, within = local & 511;
    int lane = within >> 4, e = within & 15;
    int nt = blk & 7, kcomb = blk >> 3;
    int k = kcomb >> 2, kc = kcomb & 3;
    int N = nt * 16 + (lane & 15);
    int K = kc * 32 + (e < 8 ? e : e + 8) + ((lane >> 4) << 3);
    val = Wheads[(k * 128 + N) * 128 + K];
  } else {                                 // dec
    int local = idx - PK_DEC;
    int blk = local >> 9, within = local & 511;
    int lane = within >> 4, e = within & 15;
    int nt = blk & 7, kc = blk >> 3;
    int N = nt * 16 + (lane & 15);
    int K = kc * 32 + (e < 8 ? e : e + 8) + ((lane >> 4) << 3);
    val = Wdec[N * 256 + K];
  }
  ws[idx] = (__bf16)val;
}

// -------------------- per-wave LDS layout (bytes) --------------------------
// xs   : float[16][40]   = 2560   @ 0
// obse : bf16 [16][392]  = 12544  @ 2560   (only cols 0..383 used)
// ench : bf16 [16][136]  = 4352   @ 15104  (only cols 0..127 used)
// sc   : float[16][8]    = 512    @ 19456  (scores, then attn)
#define WAVE_LDS 19968

__global__ __launch_bounds__(256, 1)
void fused_qnet(const float* __restrict__ x,
                const float* __restrict__ Wact, const float* __restrict__ bact,
                const float* __restrict__ bobs, const float* __restrict__ benc,
                const float* __restrict__ bheads, const float* __restrict__ bdec,
                const float* __restrict__ Wq, const float* __restrict__ bq,
                const int* __restrict__ agent_p,
                const __bf16* __restrict__ pack,
                float* __restrict__ out) {
  extern __shared__ char smem[];
  const int lane = threadIdx.x & 31;
  const int wave = threadIdx.x >> 5;
  const int g = lane >> 4;       // 16-lane group
  const int lm = lane & 15;
  const int tile = blockIdx.x * 8 + wave;
  const int row0 = tile * 16;

  char* wbase = smem + wave * WAVE_LDS;
  float*  xs   = (float*)(wbase);
  __bf16* obse = (__bf16*)(wbase + 2560);
  __bf16* ench = (__bf16*)(wbase + 15104);
  float*  sc   = (float*)(wbase + 19456);

  const __bf16* P_OBS   = pack + PK_OBS;
  const __bf16* P_ENC   = pack + PK_ENC;
  const __bf16* P_HEADS = pack + PK_HEADS;
  const __bf16* P_DEC   = pack + PK_DEC;

  // ---- 1. stage x tile into LDS (contiguous 16*39 floats) ----
  const float* xt = x + (size_t)row0 * 39;
  for (int i = lane; i < 16 * 39; i += 32) {
    int r = i / 39, c = i - r * 39;
    xs[r * 40 + c] = xt[i];
  }
  __syncthreads();

  const int agent = agent_p[0];
  const int o0 = (agent == 0) ? 1 : 0;
  const int o1 = (agent == 2) ? 1 : 2;

  // ---- 2. obs encodings: [16,12]@[12,128] via WMMA (K padded to 32) ----
#pragma unroll
  for (int n = 0; n < 3; ++n) {
    v16bf a;
#pragma unroll
    for (int e = 0; e < 16; ++e) {
      int kk = (e < 8 ? e : e + 8) + 8 * g;
      a[e] = (kk < 12) ? (__bf16)xs[lm * 40 + n * 12 + kk] : (__bf16)0.0f;
    }
#pragma unroll
    for (int nt = 0; nt < 8; ++nt) {
      v8f c = vzero();
      c = wmma_bf16(a, ldB(P_OBS, nt, lane), c);
      float bias = bobs[nt * 16 + lm];
#pragma unroll
      for (int r = 0; r < 8; ++r)
        obse[(r + 8 * g) * 392 + n * 128 + nt * 16 + lm] =
            (__bf16)fmaxf(c[r] + bias, 0.0f);
    }
  }
  __syncthreads();

  // ---- 3. build enc_in A fragments (K = 512 -> 16 chunks of 32) ----
  v16bf Aenc[16];
#pragma unroll
  for (int kc = 0; kc < 12; ++kc)
#pragma unroll
    for (int e = 0; e < 16; ++e) {
      int K = kc * 32 + (e < 8 ? e : e + 8) + 8 * g;
      Aenc[kc][e] = obse[lm * 392 + K];
    }
  {
    float xa = xs[lm * 40 + 36 + agent];      // own action scalar
#pragma unroll
    for (int kc = 12; kc < 16; ++kc)
#pragma unroll
      for (int e = 0; e < 16; ++e) {
        int h = (kc - 12) * 32 + (e < 8 ? e : e + 8) + 8 * g;
        Aenc[kc][e] = (__bf16)fmaxf(xa * Wact[h] + bact[h], 0.0f);
      }
  }

  // ---- 4. enc_h = relu(enc_in @ Wenc^T + benc), staged to LDS bf16 ----
#pragma unroll
  for (int nt = 0; nt < 8; ++nt) {
    v8f c = vzero();
#pragma unroll
    for (int kc = 0; kc < 16; ++kc)
      c = wmma_bf16(Aenc[kc], ldB(P_ENC, kc * 8 + nt, lane), c);
    float bias = benc[nt * 16 + lm];
#pragma unroll
    for (int r = 0; r < 8; ++r)
      ench[(r + 8 * g) * 136 + nt * 16 + lm] = (__bf16)fmaxf(c[r] + bias, 0.0f);
  }
  __syncthreads();

  // ---- 5. head A fragments from enc_h ----
  v16bf Ah[4];
#pragma unroll
  for (int kc = 0; kc < 4; ++kc)
#pragma unroll
    for (int e = 0; e < 16; ++e)
      Ah[kc][e] = ench[lm * 136 + kc * 32 + (e < 8 ? e : e + 8) + 8 * g];

  // ---- 6. dec_H = relu(dec_in @ Wdec^T + bdec); A built directly ----
  v16bf Ad[8];
  {
    float xa0 = xs[lm * 40 + 36 + o0];
    float xa1 = xs[lm * 40 + 36 + o1];
#pragma unroll
    for (int kc = 0; kc < 8; ++kc)
#pragma unroll
      for (int e = 0; e < 16; ++e) {
        int K = kc * 32 + (e < 8 ? e : e + 8) + 8 * g;
        int h = K & 127;
        float xa = (K < 128) ? xa0 : xa1;
        Ad[kc][e] = (__bf16)fmaxf(xa * Wact[h] + bact[h], 0.0f);
      }
  }
  v8f dH[8];
#pragma unroll
  for (int nt = 0; nt < 8; ++nt) {
    v8f c = vzero();
#pragma unroll
    for (int kc = 0; kc < 8; ++kc)
      c = wmma_bf16(Ad[kc], ldB(P_DEC, kc * 8 + nt, lane), c);
    float bias = bdec[nt * 16 + lm];
#pragma unroll
    for (int r = 0; r < 8; ++r) dH[nt][r] = fmaxf(c[r] + bias, 0.0f);
  }

  // ---- 7. pass 1: scores[b,k] = <head_k[b], dec_H[b]> ----
#pragma unroll
  for (int k = 0; k < 8; ++k) {
    float p[8];
#pragma unroll
    for (int r = 0; r < 8; ++r) p[r] = 0.0f;
#pragma unroll
    for (int nt = 0; nt < 8; ++nt) {
      v8f c = vzero();
#pragma unroll
      for (int kc = 0; kc < 4; ++kc)
        c = wmma_bf16(Ah[kc], ldB(P_HEADS, (k * 4 + kc) * 8 + nt, lane), c);
      float bias = bheads[k * 128 + nt * 16 + lm];
#pragma unroll
      for (int r = 0; r < 8; ++r)
        p[r] += fmaxf(c[r] + bias, 0.0f) * dH[nt][r];
    }
#pragma unroll
    for (int r = 0; r < 8; ++r) {  // reduce over h (16 lanes of group)
      float s = p[r];
      s += __shfl_xor(s, 1, 32);
      s += __shfl_xor(s, 2, 32);
      s += __shfl_xor(s, 4, 32);
      s += __shfl_xor(s, 8, 32);
      if (lm == 0) sc[(r + 8 * g) * 8 + k] = s;
    }
  }
  __syncthreads();

  // ---- 8. softmax over k (one row per lane 0..15) ----
  if (lane < 16) {
    float s[8], m = -INFINITY;
#pragma unroll
    for (int k = 0; k < 8; ++k) { s[k] = sc[lane * 8 + k]; m = fmaxf(m, s[k]); }
    float sum = 0.0f;
#pragma unroll
    for (int k = 0; k < 8; ++k) { s[k] = __expf(s[k] - m); sum += s[k]; }
    float inv = 1.0f / sum;
#pragma unroll
    for (int k = 0; k < 8; ++k) sc[lane * 8 + k] = s[k] * inv;
  }
  __syncthreads();

  // ---- 9. pass 2: ctx = sum_k attn_k * head_k (recompute heads) ----
  v8f ctx[8];
#pragma unroll
  for (int nt = 0; nt < 8; ++nt) ctx[nt] = vzero();
#pragma unroll
  for (int k = 0; k < 8; ++k) {
    float w[8];
#pragma unroll
    for (int r = 0; r < 8; ++r) w[r] = sc[(r + 8 * g) * 8 + k];
#pragma unroll
    for (int nt = 0; nt < 8; ++nt) {
      v8f c = vzero();
#pragma unroll
      for (int kc = 0; kc < 4; ++kc)
        c = wmma_bf16(Ah[kc], ldB(P_HEADS, (k * 4 + kc) * 8 + nt, lane), c);
      float bias = bheads[k * 128 + nt * 16 + lm];
#pragma unroll
      for (int r = 0; r < 8; ++r)
        ctx[nt][r] += w[r] * fmaxf(c[r] + bias, 0.0f);
    }
  }

  // ---- 10. qvalue = ctx @ Wq^T + bq ----
  float p[8];
#pragma unroll
  for (int r = 0; r < 8; ++r) p[r] = 0.0f;
#pragma unroll
  for (int nt = 0; nt < 8; ++nt) {
    float wq = Wq[nt * 16 + lm];
#pragma unroll
    for (int r = 0; r < 8; ++r) p[r] += ctx[nt][r] * wq;
  }
  float bqv = bq[0];
#pragma unroll
  for (int r = 0; r < 8; ++r) {
    float s = p[r];
    s += __shfl_xor(s, 1, 32);
    s += __shfl_xor(s, 2, 32);
    s += __shfl_xor(s, 4, 32);
    s += __shfl_xor(s, 8, 32);
    if (lm == 0) out[row0 + r + 8 * g] = s + bqv;
  }
}

extern "C" void kernel_launch(void* const* d_in, const int* in_sizes, int n_in,
                              void* d_out, int out_size, void* d_ws, size_t ws_size,
                              hipStream_t stream) {
  const float* x      = (const float*)d_in[0];
  const float* Wobs   = (const float*)d_in[1];
  const float* bobs   = (const float*)d_in[2];
  const float* Wact   = (const float*)d_in[3];
  const float* bact   = (const float*)d_in[4];
  const float* Wenc   = (const float*)d_in[5];
  const float* benc   = (const float*)d_in[6];
  const float* Wheads = (const float*)d_in[7];
  const float* bheads = (const float*)d_in[8];
  const float* Wdec   = (const float*)d_in[9];
  const float* bdec   = (const float*)d_in[10];
  const float* Wq     = (const float*)d_in[11];
  const float* bq     = (const float*)d_in[12];
  const int*   agent  = (const int*)d_in[13];

  __bf16* pack = (__bf16*)d_ws;       // needs 466944 bytes of workspace
  float*  out  = (float*)d_out;

  // 1) pack weights to bf16 in WMMA B-operand layout (tiny, one-shot per call)
  prep_pack<<<(PK_TOTAL + 255) / 256, 256, 0, stream>>>(Wobs, Wenc, Wheads, Wdec, pack);

  // 2) fused network: 65536 rows / (16 rows * 8 waves) = 512 blocks
  const int smem_bytes = 8 * WAVE_LDS;  // 159744 B (< 320 KB WGP LDS)
  fused_qnet<<<512, 256, smem_bytes, stream>>>(
      x, Wact, bact, bobs, benc, bheads, bdec, Wq, bq, agent, pack, out);
}
